// PhrasePatchContrastive_65335042507170
// MI455X (gfx1250) — compile-verified, matrix operand order checked by reading, not run on hardware
//
#include <hip/hip_runtime.h>
#include <hip/hip_bf16.h>

typedef __attribute__((ext_vector_type(16))) _Float16 v16h;
typedef __attribute__((ext_vector_type(8)))  _Float16 v8h;
typedef __attribute__((ext_vector_type(8)))  float    v8f;
typedef __attribute__((ext_vector_type(4)))  uint32_t v4u;
typedef __attribute__((ext_vector_type(8)))  uint32_t v8u;

#define DIM      128
#define NPH      8192
#define NBATCH   32
#define NPP      196
#define NPATCH   (NBATCH * NPP)   // 6272
#define PPB      (NPH / NBATCH)   // 256
#define EPS_LN   1e-5f
#define EPS_NORM 1e-12f

// ---------------------------------------------------------------------------
// TDM: async DMA of a 64-row x 128-col f16 tile (row stride 128) from global
// memory into LDS. D# per CDNA5 ISA. Tracked TENSORcnt.
// ---------------------------------------------------------------------------
__device__ __forceinline__ void tdm_load_tile64x128(const _Float16* gsrc,
                                                    uint32_t lds_off)
{
    const uint64_t ga = (uint64_t)(uintptr_t)gsrc;
    v4u g0;
    g0.x = 1u;                                            // count=1 (valid user D#)
    g0.y = lds_off;                                       // lds_addr (bytes)
    g0.z = (uint32_t)(ga & 0xffffffffu);                  // global_addr[31:0]
    g0.w = (uint32_t)((ga >> 32) & 0x01ffffffu)           // global_addr[56:32]
         | (2u << 30);                                    // type = 2 ("image")
    v8u g1 = { 1u << 16,                                  // data_size = 2 bytes
               (uint32_t)(DIM & 0xffff) << 16,            // tensor_dim0 = 128
               (uint32_t)(64 & 0xffff) << 16,             // tensor_dim1 = 64
               (uint32_t)DIM << 16,                       // tile_dim0 = 128
               64u,                                       // tile_dim1 = 64
               (uint32_t)DIM,                             // tensor_dim0_stride = 128
               0u, 0u };
    asm volatile("tensor_load_to_lds %0, %1" :: "s"(g0), "s"(g1) : "memory");
}

// Load the 4 K-chunk B fragments of one 16-col tile from LDS (ds_load_b128 x8).
__device__ __forceinline__ void load_bfr(v16h (&dst)[4], const _Float16* bp)
{
    #pragma unroll
    for (int kc = 0; kc < 4; ++kc) {
        v8h lo = *(const v8h*)(bp + kc * 32);
        v8h hi = *(const v8h*)(bp + kc * 32 + 16);
        dst[kc] = __builtin_shufflevector(lo, hi, 0,1,2,3,4,5,6,7,8,9,10,11,12,13,14,15);
    }
}

// ---------------------------------------------------------------------------
// Projection: Linear -> LayerNorm -> exact GELU -> Linear, then write fp32
// output and an L2-normalized f16 copy for the WMMA similarity GEMM.
// ---------------------------------------------------------------------------
__global__ __launch_bounds__(128) void proj_kernel(
    const float* __restrict__ X,
    const float* __restrict__ W1, const float* __restrict__ b1,
    const float* __restrict__ gam, const float* __restrict__ bet,
    const float* __restrict__ W2, const float* __restrict__ b2,
    float* __restrict__ Y, _Float16* __restrict__ Yn)
{
    const int ROWS = 4;
    __shared__ float xs[ROWS][DIM];
    __shared__ float hs[ROWS][DIM];
    __shared__ float red[DIM];

    const int j = threadIdx.x;
    const int rbase = blockIdx.x * ROWS;

    #pragma unroll
    for (int r = 0; r < ROWS; ++r)
        xs[r][j] = X[(size_t)(rbase + r) * DIM + j];
    __syncthreads();

    float acc[ROWS];
    {
        const float bb = b1[j];
        #pragma unroll
        for (int r = 0; r < ROWS; ++r) acc[r] = bb;
        for (int k = 0; k < DIM; ++k) {
            const float w = W1[k * DIM + j];
            #pragma unroll
            for (int r = 0; r < ROWS; ++r) acc[r] += xs[r][k] * w;
        }
    }

    const float gj = gam[j], bj = bet[j];
    for (int r = 0; r < ROWS; ++r) {
        red[j] = acc[r]; __syncthreads();
        for (int s = 64; s > 0; s >>= 1) { if (j < s) red[j] += red[j + s]; __syncthreads(); }
        const float mu = red[0] * (1.0f / DIM); __syncthreads();
        const float d = acc[r] - mu;
        red[j] = d * d; __syncthreads();
        for (int s = 64; s > 0; s >>= 1) { if (j < s) red[j] += red[j + s]; __syncthreads(); }
        const float var = red[0] * (1.0f / DIM); __syncthreads();
        const float hn = d * rsqrtf(var + EPS_LN) * gj + bj;
        hs[r][j] = 0.5f * hn * (1.0f + erff(hn * 0.70710678118654752f)); // exact GELU
    }
    __syncthreads();

    {
        const float bb = b2[j];
        #pragma unroll
        for (int r = 0; r < ROWS; ++r) acc[r] = bb;
        for (int k = 0; k < DIM; ++k) {
            const float w = W2[k * DIM + j];
            #pragma unroll
            for (int r = 0; r < ROWS; ++r) acc[r] += hs[r][k] * w;
        }
    }

    for (int r = 0; r < ROWS; ++r) {
        red[j] = acc[r] * acc[r]; __syncthreads();
        for (int s = 64; s > 0; s >>= 1) { if (j < s) red[j] += red[j + s]; __syncthreads(); }
        const float nrm = sqrtf(red[0]); __syncthreads();
        const float inv = 1.0f / fmaxf(nrm, EPS_NORM);
        const size_t off = (size_t)(rbase + r) * DIM + j;
        Y[off]  = acc[r];
        Yn[off] = (_Float16)(acc[r] * inv);
    }
}

// ---------------------------------------------------------------------------
// Fused similarity GEMM + both logsumexp reductions.
// TDM double-buffers 64x128 f16 B-tiles into LDS (TENSORcnt-tracked) while
// WMMAs consume the other buffer; B fragments are software-pipelined across
// n-tiles so ds_load latency hides under the WMMAs; A fragments persist in
// registers for the whole column sweep.
// ---------------------------------------------------------------------------
__global__ __launch_bounds__(128) void sim_lse_kernel(
    const _Float16* __restrict__ An, const _Float16* __restrict__ Bn,
    const float* __restrict__ temp,
    float* __restrict__ row_sumexp, float* __restrict__ col_sumexp)
{
    __shared__ _Float16 Bsh[2][64 * DIM];   // 2 x 16 KB double buffer
    __shared__ float smemcol[64];

    const float invt  = 1.0f / temp[0];
    const float SHIFT = invt;               // cosine <= 1 -> exp(sim - 1/T) in (0,1]

    const int tid  = threadIdx.x;
    const int lane = tid & 31;
    const int wave = tid >> 5;
    const int row0 = blockIdx.x * 64 + wave * 16;
    const int colg0 = blockIdx.y * 448;     // 14 groups of 7 x 64 cols
    const int r    = lane & 15;             // M (or N) within tile
    const int koff = (lane >> 4) * 8;       // K half-select (ISA 16-bit layout)

    // Persistent A fragments: the wave's 16-row strip, 4 K-chunks.
    v16h afr[4];
    const _Float16* ap = An + (size_t)(row0 + r) * DIM;
    #pragma unroll
    for (int kc = 0; kc < 4; ++kc) {
        v8h lo = *(const v8h*)(ap + kc * 32 + koff);
        v8h hi = *(const v8h*)(ap + kc * 32 + koff + 16);
        afr[kc] = __builtin_shufflevector(lo, hi, 0,1,2,3,4,5,6,7,8,9,10,11,12,13,14,15);
    }

    float rowacc[8];
    #pragma unroll
    for (int rr = 0; rr < 8; ++rr) rowacc[rr] = 0.0f;

    // Prologue: kick off DMA of chunk 0 into buffer 0 (one wave issues).
    if (wave == 0)
        tdm_load_tile64x128(Bn + (size_t)colg0 * DIM,
                            (uint32_t)(uintptr_t)&Bsh[0][0]);

    for (int chunk = 0; chunk < 7; ++chunk) {
        const int colbase = colg0 + chunk * 64;
        const int buf = chunk & 1;

        if (wave == 0) {
            if (chunk + 1 < 7) {
                tdm_load_tile64x128(Bn + (size_t)(colbase + 64) * DIM,
                                    (uint32_t)(uintptr_t)&Bsh[buf ^ 1][0]);
                __builtin_amdgcn_s_wait_tensorcnt(1);   // this chunk's tile landed
            } else {
                __builtin_amdgcn_s_wait_tensorcnt(0);   // last tile landed
            }
        }
        if (tid < 64) smemcol[tid] = 0.0f;
        __syncthreads();    // tile visible to all waves; smemcol reset

        const _Float16* bbase = &Bsh[buf][0];

        // Software pipeline: fetch n-tile nt+1's fragments before nt's WMMAs.
        v16h bfr[2][4];
        load_bfr(bfr[0], bbase + (size_t)r * DIM + koff);

        #pragma unroll
        for (int nt = 0; nt < 4; ++nt) {
            const int cur = nt & 1;
            if (nt + 1 < 4)
                load_bfr(bfr[cur ^ 1],
                         bbase + (size_t)((nt + 1) * 16 + r) * DIM + koff);

            v8f c = {};
            #pragma unroll
            for (int kc = 0; kc < 4; ++kc)
                c = __builtin_amdgcn_wmma_f32_16x16x32_f16(
                        false, afr[kc], false, bfr[cur][kc], (short)0, c, false, false);

            // C layout: VGPR rr -> M = rr (+8 if lane>=16); N = lane & 15.
            float cl = 0.0f;
            #pragma unroll
            for (int rr = 0; rr < 8; ++rr) {
                const float e = __expf(c[rr] * invt - SHIFT);
                rowacc[rr] += e;
                cl += e;
            }
            cl += __shfl_xor(cl, 16, 32);   // fold both M halves of this column
            if (lane < 16) atomicAdd(&smemcol[nt * 16 + lane], cl); // ds_add_f32
        }
        __syncthreads();    // all waves done with this buffer + smemcol final
        if (tid < 64) atomicAdd(&col_sumexp[colbase + tid], smemcol[tid]);
    }

    // Row sums: reduce across the 16 lanes of each half-wave (wave32).
    #pragma unroll
    for (int rr = 0; rr < 8; ++rr) {
        float v = rowacc[rr];
        v += __shfl_xor(v, 1, 32);
        v += __shfl_xor(v, 2, 32);
        v += __shfl_xor(v, 4, 32);
        v += __shfl_xor(v, 8, 32);
        if (lane == 0)       atomicAdd(&row_sumexp[row0 + rr], v);
        else if (lane == 16) atomicAdd(&row_sumexp[row0 + 8 + rr], v);
    }
}

// ---------------------------------------------------------------------------
// Positive logits for patch->phrase direction: sim2[j, bidx(j)*PPB].
// ---------------------------------------------------------------------------
__global__ void pos_pt2p_kernel(const _Float16* __restrict__ Pn,
                                const _Float16* __restrict__ Phn,
                                const float* __restrict__ temp,
                                float* __restrict__ pos)
{
    const int j = blockIdx.x * blockDim.x + threadIdx.x;
    if (j >= NPATCH) return;
    const int p = (j / NPP) * PPB;
    const _Float16* a = Pn  + (size_t)j * DIM;
    const _Float16* b = Phn + (size_t)p * DIM;
    float s = 0.0f;
    for (int k = 0; k < DIM; ++k) s += (float)a[k] * (float)b[k];
    pos[j] = s / temp[0];
}

// ---------------------------------------------------------------------------
// Correspondence top-1 within the phrase's batch block + phrase->patch
// positive logit (column 0 of the block). One block per phrase.
// ---------------------------------------------------------------------------
__global__ __launch_bounds__(256) void corr_kernel(
    const _Float16* __restrict__ Phn, const _Float16* __restrict__ Pn,
    const int* __restrict__ bidx, const float* __restrict__ temp,
    float* __restrict__ pos_p2pt, float* __restrict__ top_score,
    float* __restrict__ top_idx)
{
    __shared__ float xs[DIM];
    __shared__ float sc[256];
    __shared__ int   si[256];

    const int i = blockIdx.x, j = threadIdx.x;
    if (j < DIM) xs[j] = (float)Phn[(size_t)i * DIM + j];
    __syncthreads();

    const int b = bidx[i];
    float s = -1e30f;
    if (j < NPP) {
        const _Float16* pp = Pn + (size_t)(b * NPP + j) * DIM;
        float acc = 0.0f;
        for (int k = 0; k < DIM; ++k) acc += xs[k] * (float)pp[k];
        s = acc;
        if (j == 0) pos_p2pt[i] = acc / temp[0];
    }
    sc[j] = s; si[j] = j;
    __syncthreads();
    for (int st = 128; st > 0; st >>= 1) {
        if (j < st) {
            if (sc[j + st] > sc[j] || (sc[j + st] == sc[j] && si[j + st] < si[j])) {
                sc[j] = sc[j + st]; si[j] = si[j + st];
            }
        }
        __syncthreads();
    }
    if (j == 0) { top_score[i] = sc[0]; top_idx[i] = (float)si[0]; }
}

// ---------------------------------------------------------------------------
// Final losses: mean(log(sumexp) + SHIFT - pos) in both directions.
// ---------------------------------------------------------------------------
__global__ __launch_bounds__(256) void loss_kernel(
    const float* __restrict__ row_se, const float* __restrict__ col_se,
    const float* __restrict__ pos_r, const float* __restrict__ pos_c,
    const float* __restrict__ temp, float* __restrict__ out)
{
    __shared__ float r1[256], r2[256];
    const int j = threadIdx.x;
    const float shift = 1.0f / temp[0];
    float a = 0.0f, b = 0.0f;
    for (int i = j; i < NPH;    i += 256) a += __logf(row_se[i]) + shift - pos_r[i];
    for (int i = j; i < NPATCH; i += 256) b += __logf(col_se[i]) + shift - pos_c[i];
    r1[j] = a; r2[j] = b; __syncthreads();
    for (int s = 128; s > 0; s >>= 1) {
        if (j < s) { r1[j] += r1[j + s]; r2[j] += r2[j + s]; }
        __syncthreads();
    }
    if (j == 0) {
        const float l1 = r1[0] / NPH, l2 = r2[0] / NPATCH;
        out[0] = l1 + l2; out[1] = l1; out[2] = l2;
    }
}

// ---------------------------------------------------------------------------
extern "C" void kernel_launch(void* const* d_in, const int* in_sizes, int n_in,
                              void* d_out, int out_size, void* d_ws, size_t ws_size,
                              hipStream_t stream)
{
    (void)in_sizes; (void)n_in; (void)out_size; (void)ws_size;

    const float* phrase = (const float*)d_in[0];
    const float* patch  = (const float*)d_in[1];
    const int*   bidx   = (const int*)  d_in[2];
    const float* temp   = (const float*)d_in[3];
    const float* ph_W1  = (const float*)d_in[4];
    const float* ph_b1  = (const float*)d_in[5];
    const float* ph_g   = (const float*)d_in[6];
    const float* ph_be  = (const float*)d_in[7];
    const float* ph_W2  = (const float*)d_in[8];
    const float* ph_b2  = (const float*)d_in[9];
    const float* pa_W1  = (const float*)d_in[10];
    const float* pa_b1  = (const float*)d_in[11];
    const float* pa_g   = (const float*)d_in[12];
    const float* pa_be  = (const float*)d_in[13];
    const float* pa_W2  = (const float*)d_in[14];
    const float* pa_b2  = (const float*)d_in[15];

    float* out = (float*)d_out;
    char*  ws  = (char*)d_ws;

    // Workspace layout (bytes)
    _Float16* phrase_n = (_Float16*)(ws);                 // 8192*128*2 = 2,097,152
    _Float16* patch_n  = (_Float16*)(ws + 2097152);       // 6272*128*2 = 1,605,632
    float*    row_se   = (float*)(ws + 3702784);          // 8192*4
    float*    col_se   = (float*)(ws + 3735552);          // 6272*4
    float*    pos_r    = (float*)(ws + 3760640);          // 8192*4
    float*    pos_c    = (float*)(ws + 3793408);          // 6272*4

    // d_out: total, loss_p2pt, loss_pt2p, phrase_out, patch_out, top_score, top_idx
    float* out_phrase   = out + 3;
    float* out_patch    = out_phrase + (size_t)NPH * DIM;
    float* out_topscore = out_patch + (size_t)NPATCH * DIM;
    float* out_topidx   = out_topscore + NPH;

    hipMemsetAsync(row_se, 0, NPH * sizeof(float), stream);
    hipMemsetAsync(col_se, 0, NPATCH * sizeof(float), stream);

    proj_kernel<<<NPH / 4, 128, 0, stream>>>(phrase, ph_W1, ph_b1, ph_g, ph_be,
                                             ph_W2, ph_b2, out_phrase, phrase_n);
    proj_kernel<<<NPATCH / 4, 128, 0, stream>>>(patch, pa_W1, pa_b1, pa_g, pa_be,
                                                pa_W2, pa_b2, out_patch, patch_n);

    dim3 g(NPH / 64, 14);
    sim_lse_kernel<<<g, 128, 0, stream>>>(phrase_n, patch_n, temp, row_se, col_se);

    pos_pt2p_kernel<<<(NPATCH + 127) / 128, 128, 0, stream>>>(patch_n, phrase_n, temp, pos_c);
    corr_kernel<<<NPH, 256, 0, stream>>>(phrase_n, patch_n, bidx, temp,
                                         pos_r, out_topscore, out_topidx);
    loss_kernel<<<1, 256, 0, stream>>>(row_se, col_se, pos_r, pos_c, temp, out);
}
